// MultiHeadLatentAttention_64579128262957
// MI455X (gfx1250) — matrix operand order repaired and checked by reading, not compiled, not run on hardware
//
#include <hip/hip_runtime.h>
#include <hip/hip_bf16.h>

// ---------------------------------------------------------------------------
// MLA attention for MI455X (gfx1250, wave32, WMMA 16x16x32 f16, f32 accum)
// B=1, S=2048, HID=1024, H=16, DH=64, LAT=256, WINDOW=512
// All intermediates kept in f16 (fits in L2); weights pre-transposed so every
// LDS staging transfer is a contiguous 16B copy (async global->LDS).
// ---------------------------------------------------------------------------

#define S_LEN   2048
#define HID_DIM 1024
#define N_HEAD  16
#define HEAD_D  64
#define LAT_DIM 256
#define WIN     512

typedef __attribute__((ext_vector_type(16))) _Float16 v16h;
typedef __attribute__((ext_vector_type(8)))  _Float16 v8h;
typedef __attribute__((ext_vector_type(4)))  _Float16 v4h;
typedef __attribute__((ext_vector_type(8)))  float    v8f;
typedef int ai4 __attribute__((vector_size(16)));   // matches builtin's v4i

#define WMMA_F16(A_, B_, C_) \
    __builtin_amdgcn_wmma_f32_16x16x32_f16(false, (A_), false, (B_), (short)0, (C_), false, false)

#ifndef __has_builtin
#define __has_builtin(x) 0
#endif
#if __has_builtin(__builtin_amdgcn_global_load_async_to_lds_b128)
#define HAVE_ASYNC_LDS 1
#else
#define HAVE_ASYNC_LDS 0
#endif

// 16-byte global -> LDS copy (async on gfx1250 when the builtin is exposed)
__device__ __forceinline__ void cp16(const _Float16* g, _Float16* l) {
#if HAVE_ASYNC_LDS
    __builtin_amdgcn_global_load_async_to_lds_b128(
        (__attribute__((address_space(1))) ai4*)(void*)g,
        (__attribute__((address_space(3))) ai4*)(void*)l,
        0, 0);
#else
    *(uint4*)(void*)l = *(const uint4*)(const void*)g;
#endif
}

// wait for this wave's async copies, then block barrier
__device__ __forceinline__ void stage_fence() {
#if HAVE_ASYNC_LDS
#if __has_builtin(__builtin_amdgcn_s_wait_asynccnt)
    __builtin_amdgcn_s_wait_asynccnt(0);
#else
    asm volatile("s_wait_asynccnt 0" ::: "memory");
#endif
#endif
    __syncthreads();
}

// A-matrix 16x32 f16 fragment (this code's contiguous k order, not the ISA table order):
//   lane m = lane&15 ; half = lane>>4
//   elems 0..15 = K(16*half+0..15)
__device__ __forceinline__ v16h make_afrag(const _Float16* row, int koff, int half) {
    const v8h lo = *(const v8h*)(row + koff + 16 * half);
    const v8h hi = *(const v8h*)(row + koff + 16 * half + 8);
    v16h a;
#pragma unroll
    for (int i = 0; i < 8; ++i) { a[i] = lo[i]; a[i + 8] = hi[i]; }
    return a;
}

// B-matrix 32x16 f16 fragment: lane n = lane&15, elems 0..15 = K(16*half+0..15)
__device__ __forceinline__ v16h make_bfrag(const _Float16* p) {
    const v8h lo = *(const v8h*)(p);
    const v8h hi = *(const v8h*)(p + 8);
    v16h b;
#pragma unroll
    for (int i = 0; i < 8; ++i) { b[i] = lo[i]; b[i + 8] = hi[i]; }
    return b;
}

// ---------------------------------------------------------------------------
// Prep kernels (one-time): f32 -> f16, and transpose+convert for weights
// ---------------------------------------------------------------------------
__global__ __launch_bounds__(256) void cvt_f16_kernel(
    const float* __restrict__ in, _Float16* __restrict__ out, int n4)
{
    const int idx = blockIdx.x * 256 + threadIdx.x;
    if (idx >= n4) return;
    const float4 v = ((const float4*)in)[idx];
    v4h o;
    o[0] = (_Float16)v.x; o[1] = (_Float16)v.y;
    o[2] = (_Float16)v.z; o[3] = (_Float16)v.w;
    ((v4h*)out)[idx] = o;
}

// Wt[n][k] = (f16) W[k][n]   (W is [K,N] row-major)
__global__ __launch_bounds__(256) void transpose_cvt_kernel(
    const float* __restrict__ W, _Float16* __restrict__ Wt, int K, int N)
{
    const int idx = blockIdx.x * 256 + threadIdx.x;
    const int k = idx / N;
    const int n = idx - k * N;
    if (k >= K) return;
    Wt[(size_t)n * K + k] = (_Float16)W[idx];
}

// ---------------------------------------------------------------------------
// C[M,N] = (A[M,K] @ Wt[N,K]^T + bias[N]) * outScale
// A, Wt in f16; outMode: 0 = f32 C[M,N], 1 = f16 C[M,N], 2 = f16 C^T[N,M]
// grid: (N/64, M/64), block 128 = 4 waves, wave -> 32x32 tile
// ---------------------------------------------------------------------------
__global__ __launch_bounds__(128) void gemm_f16wmma(
    const _Float16* __restrict__ A, const _Float16* __restrict__ Wt,
    const float* __restrict__ bias, void* __restrict__ Cout,
    int M, int N, int K, float outScale, int outMode)
{
    __shared__ _Float16 As[64][32];   // A rows (K-contiguous)
    __shared__ _Float16 Bs[64][32];   // Wt rows (K-contiguous) == B^T tile

    const int m0 = blockIdx.y * 64;
    const int n0 = blockIdx.x * 64;
    const int tid  = threadIdx.x;
    const int lane = tid & 31;
    const int wave = tid >> 5;
    const int half = lane >> 4;
    const int ln   = lane & 15;
    const int wrow = wave >> 1;
    const int wcol = wave & 1;

    v8f acc[2][2];
#pragma unroll
    for (int i = 0; i < 2; ++i)
#pragma unroll
        for (int j = 0; j < 2; ++j)
#pragma unroll
            for (int r = 0; r < 8; ++r) acc[i][j][r] = 0.0f;

    for (int kk = 0; kk < K; kk += 32) {
        // 64 rows x 32 f16 = 256 x 16B per tile, 2 per thread per tile
#pragma unroll
        for (int i = 0; i < 2; ++i) {
            const int idx = tid + i * 128;     // 0..255
            const int row = idx >> 2, seg = idx & 3;
            cp16(A  + (size_t)(m0 + row) * K + kk + seg * 8, &As[row][seg * 8]);
            cp16(Wt + (size_t)(n0 + row) * K + kk + seg * 8, &Bs[row][seg * 8]);
        }
        stage_fence();

        const v16h a0 = make_afrag(&As[wrow * 32 + ln][0], 0, half);
        const v16h a1 = make_afrag(&As[wrow * 32 + 16 + ln][0], 0, half);
        const v16h b0 = make_bfrag(&Bs[wcol * 32 + ln][16 * half]);
        const v16h b1 = make_bfrag(&Bs[wcol * 32 + 16 + ln][16 * half]);

        acc[0][0] = WMMA_F16(a0, b0, acc[0][0]);
        acc[0][1] = WMMA_F16(a0, b1, acc[0][1]);
        acc[1][0] = WMMA_F16(a1, b0, acc[1][0]);
        acc[1][1] = WMMA_F16(a1, b1, acc[1][1]);
        __syncthreads();
    }

    // C/D layout: row = half*8 + r, col = lane&15 per 16x16 tile
#pragma unroll
    for (int ms = 0; ms < 2; ++ms)
#pragma unroll
        for (int ns = 0; ns < 2; ++ns)
#pragma unroll
            for (int r = 0; r < 8; ++r) {
                const int row = m0 + wrow * 32 + ms * 16 + half * 8 + r;
                const int col = n0 + wcol * 32 + ns * 16 + ln;
                const float val = (acc[ms][ns][r] + bias[col]) * outScale;
                if (outMode == 0) {
                    ((float*)Cout)[(size_t)row * N + col] = val;
                } else if (outMode == 1) {
                    ((_Float16*)Cout)[(size_t)row * N + col] = (_Float16)val;
                } else {
                    ((_Float16*)Cout)[(size_t)col * M + row] = (_Float16)val;
                }
            }
}

// ---------------------------------------------------------------------------
// Flash-style sliding-window causal attention (f16 in, f16 out).
// q16,k16: [S,HID]; vT16: [HID,S] (d-major); ctx16: [S,HID].
// grid: (S/64)*H blocks; block = 128 = 4 waves; wave = 16 queries.
// Q is already pre-scaled by 1/sqrt(DH) (folded into the Q projection).
// ---------------------------------------------------------------------------
__global__ __launch_bounds__(128) void mla_attention_f16(
    const _Float16* __restrict__ q16, const _Float16* __restrict__ k16,
    const _Float16* __restrict__ vT16, _Float16* __restrict__ ctx16)
{
    __shared__ _Float16 Qs[64][64];     // query tile
    __shared__ _Float16 Ks[32][64];     // key block [key][d]
    __shared__ _Float16 Vts[64][32];    // V^T block [d][key]
    __shared__ _Float16 Ps[4][16][32];  // per-wave probabilities

    const int h  = blockIdx.x & (N_HEAD - 1);
    const int qt = blockIdx.x >> 4;
    const int q0 = qt * 64;
    const int tid  = threadIdx.x;
    const int lane = tid & 31;
    const int wave = tid >> 5;
    const int half = lane >> 4;
    const int ln   = lane & 15;
    const int cb   = h * HEAD_D;

    // stage Q: 64 rows x 64 f16 = 512 x 16B
#pragma unroll
    for (int i = 0; i < 4; ++i) {
        const int idx = tid + i * 128;      // 0..511
        const int row = idx >> 3, seg = idx & 7;
        cp16(q16 + (size_t)(q0 + row) * HID_DIM + cb + seg * 8, &Qs[row][seg * 8]);
    }
    stage_fence();

    const v16h aQ0 = make_afrag(&Qs[wave * 16 + ln][0], 0,  half);  // d 0..31
    const v16h aQ1 = make_afrag(&Qs[wave * 16 + ln][0], 32, half);  // d 32..63

    v8f Oc[4];
#pragma unroll
    for (int t = 0; t < 4; ++t)
#pragma unroll
        for (int r = 0; r < 8; ++r) Oc[t][r] = 0.0f;
    float mrow[8], lrow[8];
#pragma unroll
    for (int r = 0; r < 8; ++r) { mrow[r] = -1e30f; lrow[r] = 0.0f; }

    int kb_start = q0 - WIN; if (kb_start < 0) kb_start = 0;
    const int kb_end = q0 + 64;

    for (int kb = kb_start; kb < kb_end; kb += 32) {
        // stage K (32x64) and V^T (64x32): 256 x 16B each, contiguous
#pragma unroll
        for (int i = 0; i < 2; ++i) {
            const int idx = tid + i * 128;  // 0..255
            {
                const int row = idx >> 3, seg = idx & 7;
                cp16(k16 + (size_t)(kb + row) * HID_DIM + cb + seg * 8, &Ks[row][seg * 8]);
            }
            {
                const int row = idx >> 2, seg = idx & 3;
                cp16(vT16 + (size_t)(cb + row) * S_LEN + kb + seg * 8, &Vts[row][seg * 8]);
            }
        }
        stage_fence();

        // scores S[16q,32k]: two 16-key sub-blocks, d split 0..31 / 32..63
        v8f S0, S1;
#pragma unroll
        for (int r = 0; r < 8; ++r) { S0[r] = 0.0f; S1[r] = 0.0f; }
        S0 = WMMA_F16(aQ0, make_bfrag(&Ks[ln][16 * half]),      S0);
        S0 = WMMA_F16(aQ1, make_bfrag(&Ks[ln][32 + 16 * half]), S0);
        S1 = WMMA_F16(aQ0, make_bfrag(&Ks[16 + ln][16 * half]),      S1);
        S1 = WMMA_F16(aQ1, make_bfrag(&Ks[16 + ln][32 + 16 * half]), S1);

        // online softmax per row (16-lane reductions within each half-wave)
        float alpha[8];
#pragma unroll
        for (int r = 0; r < 8; ++r) {
            const int irow = q0 + wave * 16 + half * 8 + r;
            const int j0 = kb + ln;
            const int j1 = kb + 16 + ln;
            float s0 = S0[r], s1 = S1[r];
            if (j0 > irow || (irow - j0) > WIN) s0 -= 10000.0f;
            if (j1 > irow || (irow - j1) > WIN) s1 -= 10000.0f;
            float bm = fmaxf(s0, s1);
#pragma unroll
            for (int off = 1; off < 16; off <<= 1)
                bm = fmaxf(bm, __shfl_xor(bm, off, 32));
            const float mn = fmaxf(mrow[r], bm);
            const float a_ = __expf(mrow[r] - mn);
            const float p0 = __expf(s0 - mn);
            const float p1 = __expf(s1 - mn);
            float rs = p0 + p1;
#pragma unroll
            for (int off = 1; off < 16; off <<= 1)
                rs += __shfl_xor(rs, off, 32);
            lrow[r] = lrow[r] * a_ + rs;
            mrow[r] = mn;
            alpha[r] = a_;
            Ps[wave][half * 8 + r][ln]      = (_Float16)p0;
            Ps[wave][half * 8 + r][16 + ln] = (_Float16)p1;
        }
#pragma unroll
        for (int t = 0; t < 4; ++t)
#pragma unroll
            for (int r = 0; r < 8; ++r) Oc[t][r] *= alpha[r];
        __syncthreads();

        // O += P[16,32] @ V[32,64]
        const v16h aP = make_afrag(&Ps[wave][ln][0], 0, half);
        Oc[0] = WMMA_F16(aP, make_bfrag(&Vts[0  + ln][16 * half]), Oc[0]);
        Oc[1] = WMMA_F16(aP, make_bfrag(&Vts[16 + ln][16 * half]), Oc[1]);
        Oc[2] = WMMA_F16(aP, make_bfrag(&Vts[32 + ln][16 * half]), Oc[2]);
        Oc[3] = WMMA_F16(aP, make_bfrag(&Vts[48 + ln][16 * half]), Oc[3]);
        __syncthreads();
    }

    // finalize: divide by row sums, write f16 context
#pragma unroll
    for (int r = 0; r < 8; ++r) {
        const float inv = 1.0f / lrow[r];
        const int row = q0 + wave * 16 + half * 8 + r;
#pragma unroll
        for (int t = 0; t < 4; ++t) {
            const int col = cb + t * 16 + ln;
            ctx16[(size_t)row * HID_DIM + col] = (_Float16)(Oc[t][r] * inv);
        }
    }
}

// ---------------------------------------------------------------------------
// Host-side launch
// ---------------------------------------------------------------------------
extern "C" void kernel_launch(void* const* d_in, const int* in_sizes, int n_in,
                              void* d_out, int out_size, void* d_ws, size_t ws_size,
                              hipStream_t stream) {
    (void)in_sizes; (void)n_in; (void)out_size; (void)ws_size;

    const float* hidden = (const float*)d_in[0];
    // d_in[1] = attention_mask (pure causal -10000; reproduced analytically)
    const float* Wq = (const float*)d_in[2];
    const float* bq = (const float*)d_in[3];
    const float* Wl = (const float*)d_in[4];
    const float* bl = (const float*)d_in[5];
    const float* Wk = (const float*)d_in[6];
    const float* bk = (const float*)d_in[7];
    const float* Wv = (const float*)d_in[8];
    const float* bv = (const float*)d_in[9];
    const float* Wo = (const float*)d_in[10];
    const float* bo = (const float*)d_in[11];

    _Float16* ws = (_Float16*)d_ws;
    _Float16* h16   = ws;                                  // [S,HID]
    _Float16* WqT   = h16  + (size_t)S_LEN * HID_DIM;      // [HID,HID]
    _Float16* WlT   = WqT  + (size_t)HID_DIM * HID_DIM;    // [LAT,HID]
    _Float16* WkT   = WlT  + (size_t)LAT_DIM * HID_DIM;    // [HID,LAT]
    _Float16* WvT   = WkT  + (size_t)HID_DIM * LAT_DIM;    // [HID,LAT]
    _Float16* WoT   = WvT  + (size_t)HID_DIM * LAT_DIM;    // [HID,HID]
    _Float16* q16   = WoT  + (size_t)HID_DIM * HID_DIM;    // [S,HID] (pre-scaled)
    _Float16* lat16 = q16  + (size_t)S_LEN * HID_DIM;      // [S,LAT]
    _Float16* k16   = lat16 + (size_t)S_LEN * LAT_DIM;     // [S,HID]
    _Float16* vT16  = k16  + (size_t)S_LEN * HID_DIM;      // [HID,S]
    _Float16* ctx16 = vT16 + (size_t)HID_DIM * S_LEN;      // [S,HID]
    float* out = (float*)d_out;

    const dim3 b128(128), b256(256);

    // prep: hidden -> f16, weights -> transposed f16
    cvt_f16_kernel<<<dim3((S_LEN * HID_DIM / 4) / 256), b256, 0, stream>>>(
        hidden, h16, S_LEN * HID_DIM / 4);
    transpose_cvt_kernel<<<dim3((HID_DIM * HID_DIM) / 256), b256, 0, stream>>>(Wq, WqT, HID_DIM, HID_DIM);
    transpose_cvt_kernel<<<dim3((HID_DIM * LAT_DIM) / 256), b256, 0, stream>>>(Wl, WlT, HID_DIM, LAT_DIM);
    transpose_cvt_kernel<<<dim3((LAT_DIM * HID_DIM) / 256), b256, 0, stream>>>(Wk, WkT, LAT_DIM, HID_DIM);
    transpose_cvt_kernel<<<dim3((LAT_DIM * HID_DIM) / 256), b256, 0, stream>>>(Wv, WvT, LAT_DIM, HID_DIM);
    transpose_cvt_kernel<<<dim3((HID_DIM * HID_DIM) / 256), b256, 0, stream>>>(Wo, WoT, HID_DIM, HID_DIM);

    // projections (q pre-scaled by 1/sqrt(DH) = 0.125; V written transposed)
    gemm_f16wmma<<<dim3(HID_DIM / 64, S_LEN / 64), b128, 0, stream>>>(
        h16, WqT, bq, q16, S_LEN, HID_DIM, HID_DIM, 0.125f, 1);
    gemm_f16wmma<<<dim3(LAT_DIM / 64, S_LEN / 64), b128, 0, stream>>>(
        h16, WlT, bl, lat16, S_LEN, LAT_DIM, HID_DIM, 1.0f, 1);
    gemm_f16wmma<<<dim3(HID_DIM / 64, S_LEN / 64), b128, 0, stream>>>(
        lat16, WkT, bk, k16, S_LEN, HID_DIM, LAT_DIM, 1.0f, 1);
    gemm_f16wmma<<<dim3(HID_DIM / 64, S_LEN / 64), b128, 0, stream>>>(
        lat16, WvT, bv, vT16, S_LEN, HID_DIM, LAT_DIM, 1.0f, 2);

    mla_attention_f16<<<dim3((S_LEN / 64) * N_HEAD), b128, 0, stream>>>(
        q16, k16, vT16, ctx16);

    gemm_f16wmma<<<dim3(HID_DIM / 64, S_LEN / 64), b128, 0, stream>>>(
        ctx16, WoT, bo, out, S_LEN, HID_DIM, HID_DIM, 1.0f, 0);
}
